// GATLayer_2379411882399
// MI455X (gfx1250) — matrix operand order repaired and checked by reading, not compiled, not run on hardware
//
#include <hip/hip_runtime.h>
#include <math.h>

typedef __bf16 bf16_t;
typedef __attribute__((ext_vector_type(16))) __bf16 v16bf;
typedef __attribute__((ext_vector_type(8)))  float  v8f;

#define F_IN  256
#define F_OUT 128
#define NEG_SLOPE 0.2f

// LDS layout for staged W (bf16, transposed to [col][k]):
// pitch 264 bf16 (528 B) -> 33 uint4 per row, banks fully spread.
#define W_PITCH      264
#define W_ROW_U4     33
#define W_BUF_U4     (128 * W_ROW_U4)       // 4224 uint4 per buffer (hi or lo)
#define W_LDS_BYTES  (2 * 128 * W_PITCH * 2) // 135168 bytes

// ---------------------------------------------------------------- zero init
__global__ void k_zero(float* __restrict__ out, float* __restrict__ denom,
                       unsigned* __restrict__ segmax, int n_out, int n_seg) {
    int stride = gridDim.x * blockDim.x;
    int i = blockIdx.x * blockDim.x + threadIdx.x;
    for (int j = i; j < n_out; j += stride) out[j] = 0.0f;
    for (int j = i; j < n_seg; j += stride) { denom[j] = 0.0f; segmax[j] = 0u; }
}

// ------------------------------------------------- W -> bf16 hi/lo, transposed
// Wt[n][k] = W[k][n];  hi = bf16(w), lo = bf16(w - hi)
__global__ void k_prepW(const float* __restrict__ W,
                        bf16_t* __restrict__ Whi, bf16_t* __restrict__ Wlo) {
    int i = blockIdx.x * 256 + threadIdx.x;           // 0 .. 32767
    if (i >= F_IN * F_OUT) return;
    int n = i / F_IN;
    int k = i - n * F_IN;
    float w = W[(size_t)k * F_OUT + n];
    __bf16 h = (__bf16)w;
    Whi[i] = h;
    Wlo[i] = (__bf16)(w - (float)h);
}

// ---------------------------------------------------------------- GEMM (WMMA)
// Block: 256 threads = 8 wave32. Wave w computes rows [16w,16w+16) x all 128 cols.
// K loop: 8 steps of 32. 3 bf16 WMMAs per (kstep, ntile) for split precision.
__global__ __launch_bounds__(256) void k_gemm(const float* __restrict__ x,
                                              const bf16_t* __restrict__ Whi,
                                              const bf16_t* __restrict__ Wlo,
                                              float* __restrict__ Wh, int N) {
    extern __shared__ char smem[];
    bf16_t* sW = (bf16_t*)smem;
    uint4*  s4 = (uint4*)smem;

    // Cooperative stage of W hi+lo into LDS (128 KB payload, 16B vectors).
    const uint4* g0 = (const uint4*)Whi;
    const uint4* g1 = (const uint4*)Wlo;
    for (int i = threadIdx.x; i < 2 * 8192; i += 256) {
        int b = i >> 13;
        int j = i & 8191;                  // uint4 index in tight [128][256] bf16
        int row = j >> 5;                  // 32 uint4 per tight row
        int q   = j & 31;
        uint4 v = b ? g1[j] : g0[j];
        s4[b * W_BUF_U4 + row * W_ROW_U4 + q] = v;
    }
    __syncthreads();

    int wave   = (blockIdx.x << 3) + (threadIdx.x >> 5);
    int m_base = wave << 4;
    if (m_base >= N) return;               // wave-uniform exit, EXEC stays full

    int lane = threadIdx.x & 31;
    int half = lane >> 4;                  // 0: lanes 0-15, 1: lanes 16-31
    int lr   = lane & 15;

    const float* xrow = x + (size_t)(m_base + lr) * F_IN;

    v8f acc[8] = {};

    for (int ks = 0; ks < 8; ++ks) {
        int k0 = ks << 5;
        // A fragment (16x32 bf16): elems 0-7 -> K = k0 + half*8 + {0..7}
        //                          elems 8-15 -> K = k0 + 16 + half*8 + {0..7}
        const float* p0 = xrow + k0 + half * 8;
        float4 f0 = *(const float4*)(p0);
        float4 f1 = *(const float4*)(p0 + 4);
        float4 f2 = *(const float4*)(p0 + 16);
        float4 f3 = *(const float4*)(p0 + 20);
        float fa[16] = { f0.x, f0.y, f0.z, f0.w,  f1.x, f1.y, f1.z, f1.w,
                         f2.x, f2.y, f2.z, f2.w,  f3.x, f3.y, f3.z, f3.w };
        v16bf a_hi, a_lo;
        #pragma unroll
        for (int e = 0; e < 16; ++e) {
            float f  = fa[e];
            __bf16 h = (__bf16)f;
            a_hi[e] = h;
            a_lo[e] = (__bf16)(f - (float)h);
        }

        #pragma unroll
        for (int nt = 0; nt < 8; ++nt) {
            int col = (nt << 4) + lr;      // B: lane = output column
            const uint4* bp = (const uint4*)(sW + (size_t)col * W_PITCH + k0 + half * 8);
            union Frag { uint4 u[2]; v16bf v; } bh, bl;
            bh.u[0] = bp[0];               bh.u[1] = bp[2];
            bl.u[0] = bp[W_BUF_U4];        bl.u[1] = bp[W_BUF_U4 + 2];

            acc[nt] = __builtin_amdgcn_wmma_f32_16x16x32_bf16(
                          false, a_hi, false, bh.v, (short)0, acc[nt], false, false);
            acc[nt] = __builtin_amdgcn_wmma_f32_16x16x32_bf16(
                          false, a_hi, false, bl.v, (short)0, acc[nt], false, false);
            acc[nt] = __builtin_amdgcn_wmma_f32_16x16x32_bf16(
                          false, a_lo, false, bh.v, (short)0, acc[nt], false, false);
        }
    }

    // C/D layout: VGPR r holds M = half*8 + r, N = lr (per 16-lane half)
    int mrow0 = m_base + half * 8;
    #pragma unroll
    for (int nt = 0; nt < 8; ++nt) {
        int col = (nt << 4) + lr;
        float* dst = Wh + (size_t)mrow0 * F_OUT + col;
        #pragma unroll
        for (int r = 0; r < 8; ++r)
            dst[(size_t)r * F_OUT] = acc[nt][r];
    }
}

// ------------------------------------------------ per-node attention logits
__global__ __launch_bounds__(256) void k_alpha(const float* __restrict__ Wh,
                                               const float* __restrict__ a_src,
                                               const float* __restrict__ a_dst,
                                               float* __restrict__ as,
                                               float* __restrict__ ad, int N) {
    int node = (blockIdx.x << 3) + (threadIdx.x >> 5);
    if (node >= N) return;
    int lane = threadIdx.x & 31;
    float4 wh = *(const float4*)(Wh + (size_t)node * F_OUT + lane * 4);
    float4 vs = *(const float4*)(a_src + lane * 4);
    float4 vd = *(const float4*)(a_dst + lane * 4);
    float s = wh.x * vs.x + wh.y * vs.y + wh.z * vs.z + wh.w * vs.w;
    float d = wh.x * vd.x + wh.y * vd.y + wh.z * vd.z + wh.w * vd.w;
    #pragma unroll
    for (int off = 16; off > 0; off >>= 1) {
        s += __shfl_xor(s, off, 32);
        d += __shfl_xor(d, off, 32);
    }
    if (lane == 0) { as[node] = s; ad[node] = d; }
}

// -------------------------------- edge scores + segment max (ordered-uint map)
__device__ __forceinline__ unsigned order_map(float f) {
    unsigned b = __float_as_uint(f);
    return (b & 0x80000000u) ? ~b : (b | 0x80000000u);
}
__device__ __forceinline__ float order_unmap(unsigned u) {
    unsigned b = (u & 0x80000000u) ? (u ^ 0x80000000u) : ~u;
    return __uint_as_float(b);
}

__global__ void k_edge_score(const int* __restrict__ ei,
                             const float* __restrict__ as, const float* __restrict__ ad,
                             float* __restrict__ e, unsigned* __restrict__ segmax, int E) {
    int i = blockIdx.x * 256 + threadIdx.x;
    if (i >= E) return;
    int s = ei[i], d = ei[E + i];
    float v = as[s] + ad[d];
    v = (v > 0.0f) ? v : NEG_SLOPE * v;
    e[i] = v;
    atomicMax(segmax + d, order_map(v));
}

// ------------------------------------- exp(e - segmax[dst]) and denominators
__global__ void k_edge_exp(const int* __restrict__ ei, float* __restrict__ e,
                           const unsigned* __restrict__ segmax,
                           float* __restrict__ denom, int E) {
    int i = blockIdx.x * 256 + threadIdx.x;
    if (i >= E) return;
    int d = ei[E + i];
    float m  = order_unmap(segmax[d]);
    float ex = expf(e[i] - m);
    e[i] = ex;                              // reuse buffer: now holds e_exp
    atomicAdd(denom + d, ex);
}

// ---------------------------- weighted scatter: out[dst] += alpha * Wh[src]
__global__ __launch_bounds__(256) void k_agg(const int* __restrict__ ei,
                                             const float* __restrict__ ee,
                                             const float* __restrict__ denom,
                                             const float* __restrict__ Wh,
                                             float* __restrict__ out, int E) {
    int edge = (blockIdx.x << 3) + (threadIdx.x >> 5);
    if (edge >= E) return;
    int lane = threadIdx.x & 31;
    int s = ei[edge], d = ei[E + edge];
    float alpha = ee[edge] / (denom[d] + 1e-16f);
    float4 wh = *(const float4*)(Wh + (size_t)s * F_OUT + lane * 4);
    float* o = out + (size_t)d * F_OUT + lane * 4;
    atomicAdd(o + 0, alpha * wh.x);
    atomicAdd(o + 1, alpha * wh.y);
    atomicAdd(o + 2, alpha * wh.z);
    atomicAdd(o + 3, alpha * wh.w);
}

// ---------------------------------------------------------------- final ELU
__global__ void k_elu(float* __restrict__ out, int n) {
    int stride = gridDim.x * blockDim.x;
    for (int i = blockIdx.x * blockDim.x + threadIdx.x; i < n; i += stride) {
        float v = out[i];
        out[i] = (v > 0.0f) ? v : expm1f(v);
    }
}

// ---------------------------------------------------------------------------
extern "C" void kernel_launch(void* const* d_in, const int* in_sizes, int n_in,
                              void* d_out, int out_size, void* d_ws, size_t ws_size,
                              hipStream_t stream) {
    const float* x     = (const float*)d_in[0];
    const int*   ei    = (const int*)  d_in[1];
    const float* W     = (const float*)d_in[2];
    const float* a_src = (const float*)d_in[3];
    const float* a_dst = (const float*)d_in[4];
    float* out = (float*)d_out;

    const int N = in_sizes[0] / F_IN;       // 100000
    const int E = in_sizes[1] / 2;          // 1600000

    // Workspace carve-out (256B aligned slices)
    char* ws = (char*)d_ws;
    size_t off = 0;
    auto carve = [&](size_t bytes) -> void* {
        void* p = ws + off;
        off += (bytes + 255) & ~(size_t)255;
        return p;
    };
    float*    Wh     = (float*)   carve((size_t)N * F_OUT * sizeof(float));
    bf16_t*   Whi    = (bf16_t*)  carve((size_t)F_IN * F_OUT * sizeof(bf16_t));
    bf16_t*   Wlo    = (bf16_t*)  carve((size_t)F_IN * F_OUT * sizeof(bf16_t));
    float*    as     = (float*)   carve((size_t)N * sizeof(float));
    float*    ad     = (float*)   carve((size_t)N * sizeof(float));
    unsigned* segmax = (unsigned*)carve((size_t)N * sizeof(unsigned));
    float*    denom  = (float*)   carve((size_t)N * sizeof(float));
    float*    ebuf   = (float*)   carve((size_t)E * sizeof(float));

    // 1) zero accumulator (d_out) + segment buffers
    k_zero<<<2048, 256, 0, stream>>>(out, denom, segmax, N * F_OUT, N);

    // 2) split W into bf16 hi/lo, transposed to K-contiguous
    k_prepW<<<(F_IN * F_OUT + 255) / 256, 256, 0, stream>>>(W, Whi, Wlo);

    // 3) Wh = x @ W via bf16 WMMA with hi/lo compensation
    k_gemm<<<(N + 127) / 128, 256, W_LDS_BYTES, stream>>>(x, Whi, Wlo, Wh, N);

    // 4) per-node logits
    k_alpha<<<(N + 7) / 8, 256, 0, stream>>>(Wh, a_src, a_dst, as, ad, N);

    // 5) edge scores + segment max
    k_edge_score<<<(E + 255) / 256, 256, 0, stream>>>(ei, as, ad, ebuf, segmax, E);

    // 6) exp + denominators
    k_edge_exp<<<(E + 255) / 256, 256, 0, stream>>>(ei, ebuf, segmax, denom, E);

    // 7) weighted scatter-add aggregation
    k_agg<<<(E + 7) / 8, 256, 0, stream>>>(ei, ebuf, denom, Wh, out, E);

    // 8) ELU in place
    k_elu<<<2048, 256, 0, stream>>>(out, N * F_OUT);
}
